// FeaturePropagation_24824910971382
// MI455X (gfx1250) — matrix operand order repaired and checked by reading, not compiled, not run on hardware
//
#include <hip/hip_runtime.h>

// ---------------- problem constants ----------------
#define BB      16
#define NN      1024
#define NSS     4096
#define C_IN    256
#define C_SKIP  128
#define HH      256
#define K1      (C_IN + C_SKIP)   // 384
#define M_ROWS  (BB * NSS)        // 65536

// ---------------- workspace layout (bytes) ----------------
#define OFF_IDX   0u                                   // int32 [M_ROWS*3]      786,432
#define OFF_W     (OFF_IDX + M_ROWS * 3 * 4u)          // f32   [M_ROWS*3]      786,432
#define OFF_W1T   (OFF_W + M_ROWS * 3 * 4u)            // bf16  [HH * K1]       196,608
#define OFF_W2T   (OFF_W1T + HH * K1 * 2u)             // bf16  [HH * HH]       131,072
#define OFF_A     2097152u                             // bf16  [M_ROWS * K1]   50,331,648
#define OFF_H1    (OFF_A + (size_t)M_ROWS * K1 * 2u)   // bf16  [M_ROWS * HH]   33,554,432

typedef __attribute__((ext_vector_type(16))) __bf16 v16bf;
typedef __attribute__((ext_vector_type(8)))  float  v8f;

__device__ __forceinline__ unsigned short f32_to_bf16(float f) {
    unsigned int x = __float_as_uint(f);
    // round-to-nearest-even
    unsigned int r = (x + 0x7FFFu + ((x >> 16) & 1u)) >> 16;
    return (unsigned short)r;
}

// ---------------- kernel 1: 3-NN + inverse-distance weights ----------------
// grid: BB * (NSS/256) blocks, 256 threads; one thread per fine point.
__global__ void __launch_bounds__(256) knn_kernel(const float* __restrict__ pos,
                                                  const float* __restrict__ pos_skip,
                                                  int* __restrict__ idx_out,
                                                  float* __restrict__ w_out) {
    __shared__ float sx[NN], sy[NN], sz[NN];
    const int b     = blockIdx.x / (NSS / 256);
    const int chunk = blockIdx.x % (NSS / 256);
    const float* p = pos + (size_t)b * NN * 3;
    for (int j = threadIdx.x; j < NN; j += 256) {
        sx[j] = p[j * 3 + 0];
        sy[j] = p[j * 3 + 1];
        sz[j] = p[j * 3 + 2];
    }
    __syncthreads();

    const int i = b * NSS + chunk * 256 + threadIdx.x;   // global fine-point row
    const float qx = pos_skip[i * 3 + 0];
    const float qy = pos_skip[i * 3 + 1];
    const float qz = pos_skip[i * 3 + 2];

    float d0 = 3.0e38f, d1 = 3.0e38f, d2 = 3.0e38f;
    int   i0 = 0, i1 = 0, i2 = 0;
    for (int j = 0; j < NN; ++j) {
        const float dx = qx - sx[j];
        const float dy = qy - sy[j];
        const float dz = qz - sz[j];
        const float d = dx * dx + dy * dy + dz * dz;
        if (d < d0)      { d2 = d1; i2 = i1; d1 = d0; i1 = i0; d0 = d; i0 = j; }
        else if (d < d1) { d2 = d1; i2 = i1; d1 = d;  i1 = j; }
        else if (d < d2) { d2 = d;  i2 = j; }
    }
    const float w0 = 1.0f / fmaxf(d0, 1e-16f);
    const float w1 = 1.0f / fmaxf(d1, 1e-16f);
    const float w2 = 1.0f / fmaxf(d2, 1e-16f);
    const float s  = 1.0f / (w0 + w1 + w2);
    idx_out[i * 3 + 0] = b * NN + i0;   // global coarse row
    idx_out[i * 3 + 1] = b * NN + i1;
    idx_out[i * 3 + 2] = b * NN + i2;
    w_out[i * 3 + 0] = w0 * s;
    w_out[i * 3 + 1] = w1 * s;
    w_out[i * 3 + 2] = w2 * s;
}

// ---------------- kernel 2: weights f32 -> transposed bf16 ----------------
__global__ void __launch_bounds__(256) convert_w_kernel(const float* __restrict__ W1,
                                                        const float* __restrict__ W2,
                                                        unsigned short* __restrict__ W1t,
                                                        unsigned short* __restrict__ W2t) {
    const int t = blockIdx.x * blockDim.x + threadIdx.x;
    if (t < K1 * HH) {
        const int k = t / HH, n = t % HH;           // W1 is [K1, HH] row-major
        W1t[n * K1 + k] = f32_to_bf16(W1[t]);       // W1t is [HH, K1]
    } else if (t < K1 * HH + HH * HH) {
        const int t2 = t - K1 * HH;
        const int k = t2 / HH, n = t2 % HH;
        W2t[n * HH + k] = f32_to_bf16(W2[t2]);
    }
}

// ---------------- kernel 3: interpolate + concat -> bf16 A matrix ----------------
// one block (384 threads) per output row
__global__ void __launch_bounds__(K1) gather_concat_kernel(const float* __restrict__ x,
                                                           const float* __restrict__ x_skip,
                                                           const int* __restrict__ idx,
                                                           const float* __restrict__ w,
                                                           unsigned short* __restrict__ A) {
    const int i = blockIdx.x;
    const int c = threadIdx.x;
    const int i0 = idx[i * 3 + 0], i1 = idx[i * 3 + 1], i2 = idx[i * 3 + 2];
    const float w0 = w[i * 3 + 0], w1 = w[i * 3 + 1], w2 = w[i * 3 + 2];
    float v;
    if (c < C_IN) {
        v = w0 * x[(size_t)i0 * C_IN + c]
          + w1 * x[(size_t)i1 * C_IN + c]
          + w2 * x[(size_t)i2 * C_IN + c];
    } else {
        v = x_skip[(size_t)i * C_SKIP + (c - C_IN)];
    }
    A[(size_t)i * K1 + c] = f32_to_bf16(v);
}

// ---------------- kernel 4/5: bf16 WMMA GEMM ----------------
// C[M,Nc] = relu(A[M,Kd] @ Bt[Nc,Kd]^T + bias), A/Bt bf16, accum f32.
// Each wave computes a 16M x 64N strip: 1 A-fragment is reused by 4 B-fragments
// per k-step (4 accumulators) -> 4x less A traffic than one-tile-per-wave.
// Fragment layouts per CDNA5 ISA 7.12.2:
//   A 16x32: lane half selects K-halves; each dword holds a packed K pair.
//   B 32x16 (from transposed weights): same K order as A: lanes 0-15 K=0..7,16..23, lanes 16-31 K=8..15,24..31.
//   C/D 16x16 f32: VGPR r -> M = r + 8*(lane>>4), N = lane&15.
template <int Kd, bool OUT_BF16>
__global__ void __launch_bounds__(128) gemm_bf16_kernel(const unsigned short* __restrict__ A,
                                                        const unsigned short* __restrict__ Bt,
                                                        const float* __restrict__ bias,
                                                        void* __restrict__ out,
                                                        int Nc) {
    const int lane = threadIdx.x & 31;
    const int wave = threadIdx.x >> 5;
    const int stripsN = Nc >> 6;                 // 64-column strips
    const int strip = blockIdx.x * 4 + wave;
    const int tm  = strip / stripsN;
    const int tn0 = (strip % stripsN) * 4;       // first 16-col tile of the strip

    const int mA     = tm * 16 + (lane & 15);
    const int kbaseA = (lane >> 4) * 8;
    const int kbaseB = (lane >> 4) * 16;

    const unsigned int* __restrict__ Arow = (const unsigned int*)(A + (size_t)mA * Kd);
    const unsigned int* __restrict__ Brow[4];
#pragma unroll
    for (int t = 0; t < 4; ++t) {
        const int nB = (tn0 + t) * 16 + (lane & 15);
        Brow[t] = (const unsigned int*)(Bt + (size_t)nB * Kd);
    }

    v8f acc[4] = {};
    for (int k0 = 0; k0 < Kd; k0 += 32) {
        union { v16bf v; unsigned int u[8]; } af;
#pragma unroll
        for (int j = 0; j < 4; ++j)
            af.u[j] = Arow[(k0 + kbaseA + 2 * j) >> 1];
#pragma unroll
        for (int j = 0; j < 4; ++j)
            af.u[4 + j] = Arow[(k0 + 16 + kbaseA + 2 * j) >> 1];
#pragma unroll
        for (int t = 0; t < 4; ++t) {
            union { v16bf v; unsigned int u[8]; } bf;
#pragma unroll
            for (int j = 0; j < 8; ++j)
                bf.u[j] = Brow[t][(k0 + kbaseA + (j >> 2) * 16 + 2 * (j & 3)) >> 1];
            acc[t] = __builtin_amdgcn_wmma_f32_16x16x32_bf16(
                /*neg_a=*/false, af.v, /*neg_b=*/false, bf.v,
                /*c_mod=*/(short)0, acc[t], /*reuse_a=*/false, /*reuse_b=*/false);
        }
    }

    const int rbase = tm * 16 + (lane >> 4) * 8;
#pragma unroll
    for (int t = 0; t < 4; ++t) {
        const int col = (tn0 + t) * 16 + (lane & 15);
        const float bv = bias[col];
#pragma unroll
        for (int r = 0; r < 8; ++r) {
            float v = acc[t][r] + bv;
            v = v > 0.0f ? v : 0.0f;
            const size_t o = (size_t)(rbase + r) * Nc + col;
            if (OUT_BF16) {
                ((unsigned short*)out)[o] = f32_to_bf16(v);   // h1: re-read by GEMM2, keep RT
            } else {
                __builtin_nontemporal_store(v, (float*)out + o); // final output: streaming
            }
        }
    }
}

// ---------------- kernel 6: pos_skip / batch_skip passthrough ----------------
__global__ void __launch_bounds__(256) tail_kernel(const float* __restrict__ pos_skip,
                                                   float* __restrict__ out_pos,
                                                   float* __restrict__ out_batch) {
    const int i = blockIdx.x * blockDim.x + threadIdx.x;
    if (i < M_ROWS) {
        __builtin_nontemporal_store(pos_skip[i * 3 + 0], out_pos + i * 3 + 0);
        __builtin_nontemporal_store(pos_skip[i * 3 + 1], out_pos + i * 3 + 1);
        __builtin_nontemporal_store(pos_skip[i * 3 + 2], out_pos + i * 3 + 2);
        __builtin_nontemporal_store((float)(i / NSS), out_batch + i);
    }
}

extern "C" void kernel_launch(void* const* d_in, const int* in_sizes, int n_in,
                              void* d_out, int out_size, void* d_ws, size_t ws_size,
                              hipStream_t stream) {
    const float* x        = (const float*)d_in[0];
    const float* pos      = (const float*)d_in[1];
    const float* x_skip   = (const float*)d_in[2];
    const float* pos_skip = (const float*)d_in[3];
    const float* W1       = (const float*)d_in[4];
    const float* b1       = (const float*)d_in[5];
    const float* W2       = (const float*)d_in[6];
    const float* b2       = (const float*)d_in[7];

    char* ws = (char*)d_ws;
    int*            idx  = (int*)(ws + OFF_IDX);
    float*          w    = (float*)(ws + OFF_W);
    unsigned short* W1t  = (unsigned short*)(ws + OFF_W1T);
    unsigned short* W2t  = (unsigned short*)(ws + OFF_W2T);
    unsigned short* Amat = (unsigned short*)(ws + OFF_A);
    unsigned short* h1   = (unsigned short*)(ws + OFF_H1);

    float* out_h     = (float*)d_out;
    float* out_pos   = out_h + (size_t)M_ROWS * HH;
    float* out_batch = out_pos + (size_t)M_ROWS * 3;

    // 1. 3-NN + weights
    knn_kernel<<<BB * (NSS / 256), 256, 0, stream>>>(pos, pos_skip, idx, w);
    // 2. weights -> transposed bf16
    convert_w_kernel<<<(K1 * HH + HH * HH + 255) / 256, 256, 0, stream>>>(W1, W2, W1t, W2t);
    // 3. interpolate + concat -> bf16 A [65536, 384]
    gather_concat_kernel<<<M_ROWS, K1, 0, stream>>>(x, x_skip, idx, w, Amat);
    // 4. h1 = relu(A @ W1 + b1)   -> bf16 [65536, 256]
    const int strips = (M_ROWS / 16) * (HH / 64);    // 16384 strips, 4 waves/block
    gemm_bf16_kernel<K1, true><<<strips / 4, 128, 0, stream>>>(Amat, W1t, b1, h1, HH);
    // 5. out = relu(h1 @ W2 + b2) -> f32 [65536, 256]
    gemm_bf16_kernel<HH, false><<<strips / 4, 128, 0, stream>>>(h1, W2t, b2, out_h, HH);
    // 6. pos_skip / batch_skip passthrough
    tail_kernel<<<(M_ROWS + 255) / 256, 256, 0, stream>>>(pos_skip, out_pos, out_batch);
}
